// DecoderWithSCACNNAttention_11596411699984
// MI455X (gfx1250) — matrix operand order repaired and checked by reading, not compiled, not run on hardware
//
#include <hip/hip_runtime.h>

// ---------------------------------------------------------------------------
// SCA-CNN attention decoder forward, MI455X (gfx1250, wave32, WMMA).
// All GEMMs -> v_wmma_f32_16x16x32_bf16; 16x64 tile per wave (A fragment
// reused across 4 back-to-back WMMAs), compile-time B stride (immediate
// offset loads, no 64-bit add chains), clamp+mask row padding (no EXEC
// predication on loads), speculative global_prefetch_b8.
// ---------------------------------------------------------------------------

typedef __attribute__((ext_vector_type(16))) __bf16 v16bf;
typedef __attribute__((ext_vector_type(8)))  float  v8f;
typedef __attribute__((ext_vector_type(4)))  float  f32x4;

constexpr int B_  = 8;
constexpr int T1_ = 13;
constexpr int T_  = 12;
constexpr int V_  = 20000;
constexpr int E_  = 512;
constexpr int D_  = 512;
constexpr int K_  = 512;
constexpr int C_  = 2048;
constexpr int P_  = 64;          // H*H
constexpr int X_  = E_ + C_;     // 2560
constexpr int XH_ = X_ + D_;     // 3072
constexpr int G4_ = 4 * D_;      // 2048

__device__ __forceinline__ float sigmoidf_(float x) { return 1.f / (1.f + expf(-x)); }

__device__ __forceinline__ v8f vzero8()
{
    v8f z;
#pragma unroll
    for (int j = 0; j < 8; ++j) z[j] = 0.f;
    return z;
}

// ---------------------------------------------------------------------------
// bf16 WMMA GEMM: C[M,N] (+)= A[M,Kdim] * B. One wave computes a 16x64 tile
// (4 N-subtiles): per 32-K chunk, one A fragment + 4 B fragments feed 4
// back-to-back v_wmma issues.
// BT=true  : B stored [N][Kdim], row stride LDB  -> 4x b128 row loads
// BT=false : B stored [Kdim][N], row stride LDB  -> coalesced b32 loads with
//            immediate offsets (LDB is compile-time)
// ACC=true : accumulate onto existing C.
// TAIL=true: Ntiles may not be a multiple of 4 (uniform subtile guards);
//            TAIL=false compiles the guards away entirely.
// Rows >= Mrows: loads clamped to a valid row and masked by *0.0f, stores
// guarded. blockDim=128 (4 waves), gridDim.y = batch.
// ---------------------------------------------------------------------------
template <bool BT, bool ACC, int LDB, bool TAIL>
__global__ __launch_bounds__(128)
void wmma_gemm_bf16(const float* __restrict__ A, const float* __restrict__ Bm,
                    float* __restrict__ Cm, const float* __restrict__ bias,
                    int Mrows, int Mtiles, int Ntiles, int Kdim,
                    int lda, int ldc, long long strideA, long long strideC)
{
    const int lane    = threadIdx.x & 31;
    const int wave    = blockIdx.x * 4 + (threadIdx.x >> 5);
    const int ngroups = (Ntiles + 3) >> 2;
    if (wave >= Mtiles * ngroups) return;      // uniform per wave

    A  += (long long)blockIdx.y * strideA;
    Cm += (long long)blockIdx.y * strideC;

    const int mt   = wave / ngroups;
    const int ng   = wave % ngroups;
    const int m0   = mt << 4;
    const int nt0  = ng << 2;                  // first of 4 N-subtiles
    const int n0   = nt0 << 4;
    const int lrow = lane & 15;
    const int khi  = lane >> 4;

    const bool en1 = !TAIL || (nt0 + 1 < Ntiles);
    const bool en2 = !TAIL || (nt0 + 2 < Ntiles);
    const bool en3 = !TAIL || (nt0 + 3 < Ntiles);

    // A row for this lane: clamp + multiplicative mask instead of predication
    const int   arow  = m0 + lrow;
    const float amask = (arow < Mrows) ? 1.f : 0.f;
    const int   arowc = (arow < Mrows) ? arow : (Mrows - 1);
    const float* Arow = A + (long long)arowc * lda + (khi << 3);

    // ---- optional C preload (accumulate)
    auto loadC = [&](int nn, bool en) -> v8f {
        v8f r = vzero8();
        if (ACC && en) {
#pragma unroll
            for (int j = 0; j < 8; ++j) {
                const int   m   = m0 + (khi << 3) + j;
                const int   mc  = (m < Mrows) ? m : 0;
                const float msk = (m < Mrows) ? 1.f : 0.f;
                r[j] = Cm[(long long)mc * ldc + nn + lrow] * msk;
            }
        }
        return r;
    };
    v8f acc0 = loadC(n0,      true);
    v8f acc1 = loadC(n0 + 16, en1);
    v8f acc2 = loadC(n0 + 32, en2);
    v8f acc3 = loadC(n0 + 48, en3);

    // ---- B fragment: lane = column nn+lane%16, elements 0..7 -> K krow+0..7,
    //      elements 8..15 -> K krow+16..23 (krow = k0 + khi*8)
    auto loadB = [&](int nn, int krow) -> v16bf {
        v16bf bf;
        if (BT) {
            // [N][Kdim]: row nn+lrow, K offsets krow+e and krow+16+e
            const float* Bp = Bm + (long long)(nn + lrow) * LDB + krow;
#pragma unroll
            for (int e = 0; e < 8; ++e) {
                bf[e]     = (__bf16)Bp[e];
                bf[e + 8] = (__bf16)Bp[e + 16];
            }
        } else {
            // [Kdim][N]: column nn+lrow, K rows krow+e and krow+16+e
            const float* Bp = Bm + (long long)krow * LDB + nn + lrow;
#pragma unroll
            for (int e = 0; e < 8; ++e) {
                bf[e]     = (__bf16)Bp[e * LDB];
                bf[e + 8] = (__bf16)Bp[(e + 16) * LDB];
            }
        }
        return bf;
    };

    for (int k0 = 0; k0 < Kdim; k0 += 32) {
        // speculative prefetch of the next A chunk (safe past-end on gfx1250)
        __builtin_prefetch(Arow + k0 + 32, 0, 1);

        // ---- A fragment (ISA layout): elements 0..7 -> K khi*8+0..7,
        //      elements 8..15 -> K 16+khi*8+0..7; two b128 pairs.
        const f32x4* Ap = (const f32x4*)(Arow + k0);
        const f32x4 a0 = Ap[0], a1 = Ap[1], a2 = Ap[4], a3 = Ap[5];
        v16bf af;
#pragma unroll
        for (int e = 0; e < 4; ++e) {
            af[e]      = (__bf16)(a0[e] * amask);
            af[e + 4]  = (__bf16)(a1[e] * amask);
            af[e + 8]  = (__bf16)(a2[e] * amask);
            af[e + 12] = (__bf16)(a3[e] * amask);
        }

        const int krow = k0 + (khi << 3);
        v16bf bf0, bf1, bf2, bf3;
        bf0 = loadB(n0, krow);
        if (en1) bf1 = loadB(n0 + 16, krow);
        if (en2) bf2 = loadB(n0 + 32, krow);
        if (en3) bf3 = loadB(n0 + 48, krow);

        // 4 back-to-back WMMAs sharing one A fragment
        acc0 = __builtin_amdgcn_wmma_f32_16x16x32_bf16(false, af, false, bf0,
                                                       (short)0, acc0, false, false);
        if (en1) acc1 = __builtin_amdgcn_wmma_f32_16x16x32_bf16(false, af, false, bf1,
                                                                (short)0, acc1, false, false);
        if (en2) acc2 = __builtin_amdgcn_wmma_f32_16x16x32_bf16(false, af, false, bf2,
                                                                (short)0, acc2, false, false);
        if (en3) acc3 = __builtin_amdgcn_wmma_f32_16x16x32_bf16(false, af, false, bf3,
                                                                (short)0, acc3, false, false);
    }

    // C/D layout: VGPR j -> row m0 + 8*khi + j, col = nn + lane%16
    auto storeC = [&](int nn, const v8f& acc, bool en) {
        if (!en) return;
        const float bv = bias ? bias[nn + lrow] : 0.f;
#pragma unroll
        for (int j = 0; j < 8; ++j) {
            const int m = m0 + (khi << 3) + j;
            if (m < Mrows) Cm[(long long)m * ldc + nn + lrow] = acc[j] + bv;
        }
    };
    storeC(n0,      acc0, true);
    storeC(n0 + 16, acc1, en1);
    storeC(n0 + 32, acc2, en2);
    storeC(n0 + 48, acc3, en3);
}

template <bool BT, bool ACC, int LDB, bool TAIL>
static void gemm_launch(hipStream_t s, const float* A, int lda, const float* Bm,
                        float* Cm, int ldc, const float* bias,
                        int Mrows, int Mtiles, int Ntiles, int Kdim,
                        long long sA, long long sC, int batch)
{
    const int ngroups = (Ntiles + 3) / 4;
    const int waves   = Mtiles * ngroups;
    dim3 grid((waves + 3) / 4, batch);
    wmma_gemm_bf16<BT, ACC, LDB, TAIL><<<grid, 128, 0, s>>>(A, Bm, Cm, bias, Mrows, Mtiles,
                                                            Ntiles, Kdim, lda, ldc, sA, sC);
}

// ---------------------------------------------------------------------------
// Elementwise / reduction epilogues
// ---------------------------------------------------------------------------
__global__ void vmean_kernel(const float* __restrict__ enc, float* __restrict__ Vm)
{
    int i = blockIdx.x * blockDim.x + threadIdx.x;   // over B*C
    if (i >= B_ * C_) return;
    const f32x4* p = (const f32x4*)(enc + (long long)i * P_);
    float s = 0.f;
    for (int j = 0; j < P_ / 4; ++j) {
        f32x4 v = p[j];
        s += v[0] + v[1] + v[2] + v[3];
    }
    Vm[i] = s * (1.f / P_);
}

// s_beta[b,c] = sum_k tanh(Vmean[b,c]*W_c[k] + b_c[k] + (h@W_hc)[b,k]) * W_i_hat[k] + b_i_hat
__global__ void beta_logits_kernel(const float* __restrict__ Vm, const float* __restrict__ W_c,
                                   const float* __restrict__ b_c, const float* __restrict__ hWhc,
                                   const float* __restrict__ W_i_hat, const float* __restrict__ b_i_hat,
                                   float* __restrict__ sbeta)
{
    int i = blockIdx.x * blockDim.x + threadIdx.x;   // b*C + c
    if (i >= B_ * C_) return;
    int b = i / C_;
    float vm = Vm[i];
    const float* hb = hWhc + b * K_;
    float s = 0.f;
    for (int k = 0; k < K_; ++k)
        s += tanhf(vm * W_c[k] + b_c[k] + hb[k]) * W_i_hat[k];
    sbeta[i] = s + b_i_hat[0];
}

// softmax over the batch axis (axis=0 in reference), per inner index
__global__ void softmax_b_kernel(const float* __restrict__ s, float* __restrict__ o, int inner)
{
    int c = blockIdx.x * blockDim.x + threadIdx.x;
    if (c >= inner) return;
    float mx = -3.402823466e38f;
    for (int b = 0; b < B_; ++b) mx = fmaxf(mx, s[b * inner + c]);
    float den = 0.f;
    for (int b = 0; b < B_; ++b) den += expf(s[b * inner + c] - mx);
    float inv = 1.f / den;
    for (int b = 0; b < B_; ++b) o[b * inner + c] = expf(s[b * inner + c] - mx) * inv;
}

// A_att[b][p][c] = enc[b][c][p] * beta[b][c]   (A-matrix of the att_s GEMM)
__global__ void build_Aatt_kernel(const float* __restrict__ enc, const float* __restrict__ beta,
                                  float* __restrict__ Aatt)
{
    int i = blockIdx.x * blockDim.x + threadIdx.x;   // (b*P + p)*C + c
    if (i >= B_ * P_ * C_) return;
    int c  = i % C_;
    int bp = i / C_;
    int p  = bp % P_;
    int b  = bp / P_;
    Aatt[i] = enc[((long long)(b * C_ + c)) * P_ + p] * beta[b * C_ + c];
}

// s_alpha[b,p] = sum_k tanh(pre[b,p,k] + b_s[k] + (h@W_hs)[b,k]) * W_i[k] + b_i
__global__ void alpha_logits_kernel(const float* __restrict__ pre, const float* __restrict__ b_s,
                                    const float* __restrict__ hWhs, const float* __restrict__ W_i,
                                    const float* __restrict__ b_i, float* __restrict__ salpha)
{
    int i = blockIdx.x * blockDim.x + threadIdx.x;   // b*P + p
    if (i >= B_ * P_) return;
    int b = i / P_;
    const float* pr = pre + (long long)i * K_;
    const float* hb = hWhs + b * K_;
    float s = 0.f;
    for (int k = 0; k < K_; ++k)
        s += tanhf(pr[k] + b_s[k] + hb[k]) * W_i[k];
    salpha[i] = s + b_i[0];
}

// awe[b,c] = beta[b,c] * mean_p(enc[b,c,p] * alpha[b,p])
__global__ void awe_kernel(const float* __restrict__ enc, const float* __restrict__ beta,
                           const float* __restrict__ alpha, float* __restrict__ awe)
{
    int i = blockIdx.x * blockDim.x + threadIdx.x;   // b*C + c
    if (i >= B_ * C_) return;
    int b = i / C_;
    const float* e  = enc + (long long)i * P_;
    const float* al = alpha + b * P_;
    float s = 0.f;
    for (int p = 0; p < P_; ++p) s += e[p] * al[p];
    awe[i] = beta[i] * s * (1.f / P_);
}

// xh[b] = concat(emb[caps[b,t]], awe[b], h[b])  -> (B, 3072)
__global__ void build_xh_kernel(const int* __restrict__ caps, const float* __restrict__ emb,
                                const float* __restrict__ awe, const float* __restrict__ h,
                                float* __restrict__ xh, int t)
{
    int i = blockIdx.x * blockDim.x + threadIdx.x;   // b*XH + j
    if (i >= B_ * XH_) return;
    int b = i / XH_;
    int j = i % XH_;
    float v;
    if (j < E_) {
        int tok = caps[b * T1_ + t];
        v = emb[(long long)tok * E_ + j];
    } else if (j < X_) {
        v = awe[b * C_ + (j - E_)];
    } else {
        v = h[b * D_ + (j - X_)];
    }
    xh[i] = v;
}

// LSTM cell epilogue (biases fused here; gates GEMMs are bias-free)
__global__ void lstm_kernel(const float* __restrict__ gates, const float* __restrict__ b_ih,
                            const float* __restrict__ b_hh, float* __restrict__ h,
                            float* __restrict__ c)
{
    int i = blockIdx.x * blockDim.x + threadIdx.x;   // b*D + d
    if (i >= B_ * D_) return;
    int b = i / D_, d = i % D_;
    const float* g = gates + b * G4_;
    float ig = g[d]          + b_ih[d]          + b_hh[d];
    float fg = g[D_ + d]     + b_ih[D_ + d]     + b_hh[D_ + d];
    float gg = g[2 * D_ + d] + b_ih[2 * D_ + d] + b_hh[2 * D_ + d];
    float og = g[3 * D_ + d] + b_ih[3 * D_ + d] + b_hh[3 * D_ + d];
    float cn = sigmoidf_(fg) * c[i] + sigmoidf_(ig) * tanhf(gg);
    c[i] = cn;
    h[i] = sigmoidf_(og) * tanhf(cn);
}

// Tail of the output tuple: encoded_captions, decode_lengths, alphas(0), sort_ind
__global__ void tail_kernel(const int* __restrict__ caps, const int* __restrict__ cap_len,
                            float* __restrict__ out)
{
    const int capN    = B_ * T1_;        // 104
    const int base    = B_ * T_ * V_;    // predictions size
    const int alphasN = B_ * T_ * P_;    // 6144
    const int total   = capN + B_ + alphasN + B_;
    int i = blockIdx.x * blockDim.x + threadIdx.x;
    if (i >= total) return;
    if (i < capN)                        out[base + i] = (float)caps[i];
    else if (i < capN + B_)              out[base + i] = (float)(cap_len[i - capN] - 1);
    else if (i < capN + B_ + alphasN)    out[base + i] = 0.f;
    else                                 out[base + i] = (float)(i - (capN + B_ + alphasN));
}

// ---------------------------------------------------------------------------
extern "C" void kernel_launch(void* const* d_in, const int* in_sizes, int n_in,
                              void* d_out, int out_size, void* d_ws, size_t ws_size,
                              hipStream_t stream)
{
    const float* enc      = (const float*)d_in[0];
    const int*   caps     = (const int*)  d_in[1];
    const int*   cap_len  = (const int*)  d_in[2];
    const float* W_c      = (const float*)d_in[3];
    const float* W_hc     = (const float*)d_in[4];
    const float* W_i_hat  = (const float*)d_in[5];
    const float* b_c      = (const float*)d_in[6];
    const float* b_i_hat  = (const float*)d_in[7];
    const float* W_s      = (const float*)d_in[8];
    const float* W_hs     = (const float*)d_in[9];
    const float* W_i      = (const float*)d_in[10];
    const float* b_s      = (const float*)d_in[11];
    const float* b_i      = (const float*)d_in[12];
    const float* emb      = (const float*)d_in[13];
    const float* W_ih     = (const float*)d_in[14];
    const float* W_hh     = (const float*)d_in[15];
    const float* b_ih     = (const float*)d_in[16];
    const float* b_hh     = (const float*)d_in[17];
    const float* W_init_h = (const float*)d_in[18];
    const float* b_init_h = (const float*)d_in[19];
    const float* W_init_c = (const float*)d_in[20];
    const float* b_init_c = (const float*)d_in[21];
    const float* W_fc     = (const float*)d_in[22];
    const float* b_fc     = (const float*)d_in[23];
    float* out = (float*)d_out;

    // workspace carve-out (~5.7 MB f32); every buffer is written before read
    float* w = (float*)d_ws;
    float* Vmean = w;  w += B_ * C_;
    float* h     = w;  w += B_ * D_;
    float* c     = w;  w += B_ * D_;
    float* hWhc  = w;  w += B_ * K_;
    float* hWhs  = w;  w += B_ * K_;
    float* sbuf  = w;  w += B_ * C_;
    float* beta  = w;  w += B_ * C_;
    float* Aatt  = w;  w += B_ * P_ * C_;
    float* pre   = w;  w += B_ * P_ * K_;
    float* sal   = w;  w += B_ * P_;
    float* alpha = w;  w += B_ * P_;
    float* awe   = w;  w += B_ * C_;
    float* xh    = w;  w += B_ * XH_;
    float* gates = w;  w += B_ * G4_;

    auto blk = [](int n) { return dim3((n + 255) / 256); };

    // ---- init: Vmean, h0, c0, static outputs
    vmean_kernel<<<blk(B_ * C_), 256, 0, stream>>>(enc, Vmean);
    // h0 = Vmean @ W_init_h.T + b_init_h  (W_init_h is (D, C) -> BT, LDB=C)
    gemm_launch<true, false, C_, false>(stream, Vmean, C_, W_init_h, h, D_, b_init_h,
                                        B_, 1, D_ / 16, C_, 0, 0, 1);
    gemm_launch<true, false, C_, false>(stream, Vmean, C_, W_init_c, c, D_, b_init_c,
                                        B_, 1, D_ / 16, C_, 0, 0, 1);
    tail_kernel<<<blk(B_ * T1_ + B_ + B_ * T_ * P_ + B_), 256, 0, stream>>>(caps, cap_len, out);

    // ---- scan over T=12 steps
    for (int t = 0; t < T_; ++t) {
        // h @ W_hc, h @ W_hs   (W stored [512 inner][512 out] -> BT=false, LDB=512)
        gemm_launch<false, false, D_, false>(stream, h, D_, W_hc, hWhc, K_, nullptr,
                                             B_, 1, K_ / 16, D_, 0, 0, 1);
        gemm_launch<false, false, D_, false>(stream, h, D_, W_hs, hWhs, K_, nullptr,
                                             B_, 1, K_ / 16, D_, 0, 0, 1);

        // channel attention beta (rank-1 broadcast -> VALU tanh-dot), softmax over batch
        beta_logits_kernel<<<blk(B_ * C_), 256, 0, stream>>>(Vmean, W_c, b_c, hWhc,
                                                             W_i_hat, b_i_hat, sbuf);
        softmax_b_kernel<<<blk(C_), 256, 0, stream>>>(sbuf, beta, C_);

        // spatial attention: pre[b] = (cw[b]^T) @ W_s  — batched WMMA GEMM 64x512x2048
        build_Aatt_kernel<<<blk(B_ * P_ * C_), 256, 0, stream>>>(enc, beta, Aatt);
        gemm_launch<false, false, K_, false>(stream, Aatt, C_, W_s, pre, K_, nullptr,
                                             P_, P_ / 16, K_ / 16, C_,
                                             (long long)P_ * C_, (long long)P_ * K_, B_);
        alpha_logits_kernel<<<blk(B_ * P_), 256, 0, stream>>>(pre, b_s, hWhs, W_i, b_i, sal);
        softmax_b_kernel<<<blk(P_), 256, 0, stream>>>(sal, alpha, P_);
        awe_kernel<<<blk(B_ * C_), 256, 0, stream>>>(enc, beta, alpha, awe);

        // LSTM gates: [x_t, awe] @ W_ih.T  (+)= h @ W_hh.T
        build_xh_kernel<<<blk(B_ * XH_), 256, 0, stream>>>(caps, emb, awe, h, xh, t);
        gemm_launch<true, false, X_, false>(stream, xh, XH_, W_ih, gates, G4_, nullptr,
                                            B_, 1, G4_ / 16, X_, 0, 0, 1);
        gemm_launch<true, true, D_, false>(stream, h, D_, W_hh, gates, G4_, nullptr,
                                           B_, 1, G4_ / 16, D_, 0, 0, 1);
        lstm_kernel<<<blk(B_ * D_), 256, 0, stream>>>(gates, b_ih, b_hh, h, c);

        // vocabulary projection straight into d_out slice t, bias fused
        // (1250 N-tiles -> TAIL=true handles the partial 4-tile group)
        gemm_launch<true, false, D_, true>(stream, h, D_, W_fc, out + t * V_, T_ * V_, b_fc,
                                           B_, 1, V_ / 16, D_, 0, 0, 1);
    }
}
